// SpatialAttention_40037685133372
// MI455X (gfx1250) — matrix operand order repaired and checked by reading, compiled, not run on hardware
//
#include <hip/hip_runtime.h>

// ---------------------------------------------------------------------------
// SpatialAttention (graph-masked MHA + residual + LayerNorm) for gfx1250.
// bf16 WMMA (v_wmma_f32_16x16x32_bf16) everywhere; flash-style online softmax
// with transposed score tiles so softmax needs only one shfl_xor(16) and the
// P-tile transpose is two ds_store_b128 / two ds_load_b128 per wave.
// Sparsity gate: ~0.4% edge density means ~98% of 16x32 score tiles are fully
// masked -> one mask-dword load + wave ballot skips the whole tile (no WMMA,
// no K/V loads, no exp). Deterministic (mask-only dependent).
// ---------------------------------------------------------------------------

typedef __bf16 bf16;
typedef __attribute__((ext_vector_type(16))) __bf16 v16bf;
typedef __attribute__((ext_vector_type(8)))  __bf16 v8bf;
typedef __attribute__((ext_vector_type(8)))  float  v8f;

#define NNODES 4096
#define DMODEL 256
#define NHEADS 8
#define HDIM   32
#define MASK_WORDS (NNODES * (NNODES / 32))   // 524288 dwords = 2 MB bitmask

__device__ __forceinline__ bf16 f2bf(float f) {
  unsigned u = __builtin_bit_cast(unsigned, f);
  u += 0x7FFFu + ((u >> 16) & 1u);            // round-to-nearest-even
  unsigned short h = (unsigned short)(u >> 16);
  return __builtin_bit_cast(bf16, h);
}

__device__ __forceinline__ v8f wmma_bf16(v16bf a, v16bf b, v8f c) {
  // D(16x16 f32) = A(16x32 bf16) * B(32x16 bf16) + C
  return __builtin_amdgcn_wmma_f32_16x16x32_bf16(
      /*neg_a=*/false, a, /*neg_b=*/false, b,
      /*c_mod=*/(short)0, c, /*reuse_a=*/false, /*reuse_b=*/false);
}

// A-fragment (16x32 bf16): lane holds row M=(lane&15); element e -> K=e+16*hi.
// Two contiguous 16B chunks at +0 and +8 elements.
__device__ __forceinline__ v16bf ld_afrag(const bf16* chunk0) {
  v8bf a = *(const v8bf*)(chunk0);
  v8bf b = *(const v8bf*)(chunk0 + 8);
  v16bf r;
#pragma unroll
  for (int i = 0; i < 8; ++i) { r[i] = a[i]; r[i + 8] = b[i]; }
  return r;
}

// B-fragment (32x16 bf16): lane holds col N=(lane&15); element e -> K=e+16*hi.
// One contiguous 32B chunk per lane.
__device__ __forceinline__ v16bf ld_bfrag(const bf16* p) {
  v8bf a = *(const v8bf*)(p);
  v8bf b = *(const v8bf*)(p + 8);
  v16bf r;
#pragma unroll
  for (int i = 0; i < 8; ++i) { r[i] = a[i]; r[i + 8] = b[i]; }
  return r;
}

// ---------------------------------------------------------------------------
__global__ void k_zero(unsigned* w, int n) {
  int i = blockIdx.x * blockDim.x + threadIdx.x;
  if (i < n) w[i] = 0u;
}

__global__ void k_build_mask(const long long* e, int E, unsigned* mask) {
  int t = blockIdx.x * blockDim.x + threadIdx.x;
  if (t >= E) return;
  int s = (int)e[t];        // query row
  int d = (int)e[E + t];    // key col
  atomicOr(&mask[s * (NNODES / 32) + (d >> 5)], 1u << (d & 31));
}

__global__ void k_f32_to_bf16(const float* x, bf16* y, int n) {
  int i = blockIdx.x * blockDim.x + threadIdx.x;
  if (i < n) y[i] = f2bf(x[i]);
}

// Wt[o][i] = bf16(W[i][o])  (transposed so GEMM B-fragments are contiguous)
__global__ void k_wt_transpose(const float* w, bf16* wt) {
  int t = blockIdx.x * blockDim.x + threadIdx.x;   // 65536
  int o = t >> 8, i = t & 255;
  wt[t] = f2bf(w[i * DMODEL + o]);
}

// ---------------------------------------------------------------------------
// out[n][dout] = sum_i xb[n][i] * Wt[dout][i] + bias[dout]
// mode 0: store row-major bf16 [N][D]; mode 1: store transposed bf16 [D][N]
__global__ void __launch_bounds__(128)
k_gemm_qkv(const bf16* __restrict__ xb, const bf16* __restrict__ wt,
           const float* __restrict__ bias, bf16* __restrict__ out, int mode) {
  int lane = threadIdx.x & 31;
  int wave = threadIdx.x >> 5;
  int tile = blockIdx.x * 4 + wave;               // 4096 tiles (256 m x 16 d)
  int m0 = (tile >> 4) * 16, d0 = (tile & 15) * 16;
  int col = lane & 15;
  int hi = lane >> 4;

  v8f c = {};
  const bf16* arow = xb + (m0 + col) * DMODEL + 16 * hi;
  const bf16* brow = wt + (d0 + col) * DMODEL + 16 * hi;
#pragma unroll
  for (int k0 = 0; k0 < DMODEL; k0 += 32) {
    v16bf a = ld_afrag(arow + k0);
    v16bf b = ld_bfrag(brow + k0);
    c = wmma_bf16(a, b, c);
  }
  float bv = bias[d0 + col];
  if (mode == 0) {
#pragma unroll
    for (int v = 0; v < 8; ++v)
      out[(m0 + v + 8 * hi) * DMODEL + d0 + col] = f2bf(c[v] + bv);
  } else {
    v8bf pk;
#pragma unroll
    for (int v = 0; v < 8; ++v) pk[v] = f2bf(c[v] + bv);
    *(v8bf*)(out + (d0 + col) * NNODES + m0 + 8 * hi) = pk;   // Vt[d][n]
  }
}

// ---------------------------------------------------------------------------
// Flash attention per (head, 16-query tile). Transposed score tiles:
// T = K_tile(16x32) x Q^T(32x16): rows = keys (v-elements), cols = queries (lanes).
__global__ void __launch_bounds__(128)
k_attn(const bf16* __restrict__ Qb, const bf16* __restrict__ Kb,
       const bf16* __restrict__ Vt, const unsigned* __restrict__ mask,
       bf16* __restrict__ attn_out) {
  __shared__ __align__(16) bf16 lds_p[4 * 16 * 32];     // per-wave P tiles
  int lane = threadIdx.x & 31;
  int wave = threadIdx.x >> 5;
  bf16* pbuf = lds_p + wave * (16 * 32);

  int task = blockIdx.x * 4 + wave;                     // 2048 = 8 heads * 256
  int h = task >> 8;
  int m0 = (task & 255) * 16;
  int col = lane & 15;
  int hi = lane >> 4;
  const float NEG_INF = -__builtin_inff();
  const float scale = 0.17677669529663688f;             // 1/sqrt(32)

  // Q B-fragment: B[d][m] = Q[m0+m][h*32+d] (contiguous per lane), hoisted.
  v16bf qf = ld_bfrag(Qb + (m0 + col) * DMODEL + h * HDIM + 16 * hi);

  v8f acc0 = {}, acc1 = {};                              // d 0..15 / 16..31
  float mrun = NEG_INF, lrun = 0.f;

  const unsigned* mrow = mask + (m0 + col) * (NNODES / 32);
  const bf16* krow = Kb + h * HDIM + 16 * hi;
  const bf16* v0b = Vt + (h * HDIM + col) * NNODES + 16 * hi;
  const bf16* v1b = Vt + (h * HDIM + 16 + col) * NNODES + 16 * hi;

  for (int n0 = 0; n0 < NNODES; n0 += 32) {
    unsigned w = mrow[n0 >> 5];                         // 32 key-bits for query col

    // Sparsity gate: if no lane's query has any edge to these 32 keys, the
    // whole 16x32 tile contributes exactly zero -> skip WMMAs/loads/softmax.
    if (!__any(w != 0u)) continue;

    v8f t0 = {}, t1 = {};
    t0 = wmma_bf16(ld_afrag(krow + (n0 + col) * DMODEL), qf, t0);
    t1 = wmma_bf16(ld_afrag(krow + (n0 + 16 + col) * DMODEL), qf, t1);

    float p0[8], p1[8];
    float tmax = NEG_INF;
#pragma unroll
    for (int v = 0; v < 8; ++v) {
      int nl = v + 8 * hi;
      float s0 = ((w >> nl) & 1u) ? t0[v] * scale : NEG_INF;
      float s1 = ((w >> (16 + nl)) & 1u) ? t1[v] * scale : NEG_INF;
      p0[v] = s0; p1[v] = s1;
      tmax = fmaxf(tmax, fmaxf(s0, s1));
    }
    tmax = fmaxf(tmax, __shfl_xor(tmax, 16));           // both key-halves of col
    float mnew = fmaxf(mrun, tmax);
    float alpha = (mrun == NEG_INF) ? 0.f : __expf(mrun - mnew);

    float psum = 0.f;
#pragma unroll
    for (int v = 0; v < 8; ++v) {
      float e0 = (p0[v] == NEG_INF) ? 0.f : __expf(p0[v] - mnew);
      float e1 = (p1[v] == NEG_INF) ? 0.f : __expf(p1[v] - mnew);
      p0[v] = e0; p1[v] = e1;
      psum += e0 + e1;
    }
    psum += __shfl_xor(psum, 16);
    lrun = lrun * alpha + psum;
    mrun = mnew;

    // Rescale accumulators: row M = v + 8*hi; its alpha lives in lane M.
#pragma unroll
    for (int v = 0; v < 8; ++v) {
      float av = __shfl(alpha, v + 8 * hi);
      acc0[v] *= av;
      acc1[v] *= av;
    }

    // P tile (16 queries x 32 keys) to LDS: per lane 8 contiguous bf16 per subtile.
    v8bf pk0, pk1;
#pragma unroll
    for (int v = 0; v < 8; ++v) { pk0[v] = f2bf(p0[v]); pk1[v] = f2bf(p1[v]); }
    *(v8bf*)(pbuf + col * 32 + 8 * hi) = pk0;           // keys 8*hi..+7
    *(v8bf*)(pbuf + col * 32 + 16 + 8 * hi) = pk1;      // keys 16+8*hi..+7
    // DS ops are in-order within a wave; region is wave-private -> no barrier.
    v16bf pa = ld_afrag(pbuf + col * 32 + 16 * hi);     // A[m][k=n]

    v16bf vb0 = ld_bfrag(v0b + n0);                     // B[n][d] from Vt
    v16bf vb1 = ld_bfrag(v1b + n0);
    acc0 = wmma_bf16(pa, vb0, acc0);
    acc1 = wmma_bf16(pa, vb1, acc1);
  }

#pragma unroll
  for (int v = 0; v < 8; ++v) {
    float linv = 1.0f / __shfl(lrun, v + 8 * hi);       // row's softmax denom
    int row = m0 + v + 8 * hi;
    attn_out[row * DMODEL + h * HDIM + col]      = f2bf(acc0[v] * linv);
    attn_out[row * DMODEL + h * HDIM + 16 + col] = f2bf(acc1[v] * linv);
  }
}

// ---------------------------------------------------------------------------
// y = x + attended @ Wo + bo    (f32 out for the LayerNorm pass)
__global__ void __launch_bounds__(128)
k_outproj(const bf16* __restrict__ ab, const bf16* __restrict__ wot,
          const float* __restrict__ bo, const float* __restrict__ x,
          float* __restrict__ y) {
  int lane = threadIdx.x & 31;
  int wave = threadIdx.x >> 5;
  int tile = blockIdx.x * 4 + wave;
  int m0 = (tile >> 4) * 16, d0 = (tile & 15) * 16;
  int col = lane & 15;
  int hi = lane >> 4;

  v8f c = {};
  const bf16* arow = ab + (m0 + col) * DMODEL + 16 * hi;
  const bf16* brow = wot + (d0 + col) * DMODEL + 16 * hi;
#pragma unroll
  for (int k0 = 0; k0 < DMODEL; k0 += 32)
    c = wmma_bf16(ld_afrag(arow + k0), ld_bfrag(brow + k0), c);

  float bv = bo[d0 + col];
#pragma unroll
  for (int v = 0; v < 8; ++v) {
    int row = m0 + v + 8 * hi;
    y[row * DMODEL + d0 + col] = x[row * DMODEL + d0 + col] + c[v] + bv;
  }
}

// ---------------------------------------------------------------------------
__global__ void __launch_bounds__(256)
k_ln(const float* __restrict__ y, const float* __restrict__ gamma,
     const float* __restrict__ beta, float* __restrict__ out) {
  int lane = threadIdx.x & 31;
  int wave = threadIdx.x >> 5;
  int row = blockIdx.x * 8 + wave;                      // one wave per row
  const float* r = y + row * DMODEL + lane * 8;
  float v[8];
  float s = 0.f;
#pragma unroll
  for (int j = 0; j < 8; ++j) { v[j] = r[j]; s += v[j]; }
#pragma unroll
  for (int m = 16; m >= 1; m >>= 1) s += __shfl_xor(s, m);
  float mu = s * (1.0f / DMODEL);
  float q = 0.f;
#pragma unroll
  for (int j = 0; j < 8; ++j) { float d = v[j] - mu; q += d * d; }
#pragma unroll
  for (int m = 16; m >= 1; m >>= 1) q += __shfl_xor(q, m);
  float rinv = rsqrtf(q * (1.0f / DMODEL) + 1e-5f);
  float* o = out + row * DMODEL + lane * 8;
#pragma unroll
  for (int j = 0; j < 8; ++j) {
    int c0 = lane * 8 + j;
    o[j] = (v[j] - mu) * rinv * gamma[c0] + beta[c0];
  }
}

// ---------------------------------------------------------------------------
extern "C" void kernel_launch(void* const* d_in, const int* in_sizes, int n_in,
                              void* d_out, int out_size, void* d_ws, size_t ws_size,
                              hipStream_t stream) {
  const float*     x     = (const float*)d_in[0];
  const long long* edges = (const long long*)d_in[1];
  const float* Wq = (const float*)d_in[2];  const float* bq = (const float*)d_in[3];
  const float* Wk = (const float*)d_in[4];  const float* bk = (const float*)d_in[5];
  const float* Wv = (const float*)d_in[6];  const float* bv = (const float*)d_in[7];
  const float* Wo = (const float*)d_in[8];  const float* bo = (const float*)d_in[9];
  const float* gamma = (const float*)d_in[10];
  const float* beta  = (const float*)d_in[11];
  int E = in_sizes[1] / 2;                               // 69632

  // Workspace layout (~17 MB):
  char* ws = (char*)d_ws;
  unsigned* mask = (unsigned*)ws;                        // 2 MB bitmask
  bf16* xb  = (bf16*)(ws + (size_t)(2 << 20));           // 2 MB
  bf16* wqt = (bf16*)(ws + (size_t)(4 << 20));           // 4 x 128 KB
  bf16* wkt = wqt + 65536;
  bf16* wvt = wkt + 65536;
  bf16* wot = wvt + 65536;
  bf16* Qb  = (bf16*)(ws + (size_t)(5 << 20));           // 2 MB  [N][D]
  bf16* Kb  = (bf16*)(ws + (size_t)(7 << 20));           // 2 MB  [N][D]
  bf16* Vt  = (bf16*)(ws + (size_t)(9 << 20));           // 2 MB  [D][N]
  bf16* Ab  = (bf16*)(ws + (size_t)(11 << 20));          // 2 MB  [N][D]
  float* y  = (float*)(ws + (size_t)(13 << 20));         // 4 MB  [N][D]

  k_zero<<<MASK_WORDS / 256, 256, 0, stream>>>(mask, MASK_WORDS);
  k_build_mask<<<(E + 255) / 256, 256, 0, stream>>>(edges, E, mask);
  k_f32_to_bf16<<<(NNODES * DMODEL) / 256, 256, 0, stream>>>(x, xb, NNODES * DMODEL);
  k_wt_transpose<<<256, 256, 0, stream>>>(Wq, wqt);
  k_wt_transpose<<<256, 256, 0, stream>>>(Wk, wkt);
  k_wt_transpose<<<256, 256, 0, stream>>>(Wv, wvt);
  k_wt_transpose<<<256, 256, 0, stream>>>(Wo, wot);
  k_gemm_qkv<<<1024, 128, 0, stream>>>(xb, wqt, bq, Qb, 0);
  k_gemm_qkv<<<1024, 128, 0, stream>>>(xb, wkt, bk, Kb, 0);
  k_gemm_qkv<<<1024, 128, 0, stream>>>(xb, wvt, bv, Vt, 1);
  k_attn<<<512, 128, 0, stream>>>(Qb, Kb, Vt, mask, Ab);
  k_outproj<<<1024, 128, 0, stream>>>(Ab, wot, bo, x, y);
  k_ln<<<512, 256, 0, stream>>>(y, gamma, beta, (float*)d_out);
}
